// GNNSubgraphClassifier_59176059404815
// MI455X (gfx1250) — matrix operand order repaired and checked by reading, not compiled, not run on hardware
//
#include <hip/hip_runtime.h>
#include <hip/hip_bf16.h>

typedef __attribute__((ext_vector_type(16))) __bf16 v16bf;
typedef __attribute__((ext_vector_type(8)))  __bf16 v8bf;
typedef __attribute__((ext_vector_type(8)))  float  v8f;

#define CH 128   // C_in == H == 128

// ---------------- small streaming kernels ----------------

__global__ void k_fill(float* __restrict__ p, float v, int n) {
    int i = blockIdx.x * blockDim.x + threadIdx.x;
    if (i < n) p[i] = v;
}

__global__ void k_deg(const long long* __restrict__ dst, float* __restrict__ deg, int E) {
    int e = blockIdx.x * blockDim.x + threadIdx.x;
    if (e < E) unsafeAtomicAdd(&deg[(int)dst[e]], 1.0f);
}

__global__ void k_dinv(const float* __restrict__ deg, float* __restrict__ dinv, int n) {
    int i = blockIdx.x * blockDim.x + threadIdx.x;
    if (i < n) { float d = deg[i]; dinv[i] = d > 0.f ? rsqrtf(d) : 0.f; }
}

__global__ void k_cvt(const float* __restrict__ in, __bf16* __restrict__ out, int n) {
    int i = blockIdx.x * blockDim.x + threadIdx.x;
    if (i < n) out[i] = (__bf16)in[i];
}

__global__ void k_relu_cvt(const float* __restrict__ in, __bf16* __restrict__ out, int n) {
    int i = blockIdx.x * blockDim.x + threadIdx.x;
    if (i < n) out[i] = (__bf16)fmaxf(in[i], 0.f);
}

// out[i,c] = bias[c] + t[i,c] * dinv[i]^2   (self-loop message + bias)
__global__ void k_agg_init(float* __restrict__ out, const float* __restrict__ t,
                           const float* __restrict__ dinv, const float* __restrict__ bias,
                           int total) {
    int i = blockIdx.x * blockDim.x + threadIdx.x;
    if (i >= total) return;
    int node = i >> 7, c = i & (CH - 1);
    float di = dinv[node];
    out[i] = bias[c] + t[i] * di * di;
}

// one wave (32 lanes x 4 channels) per edge: out[dst] += t[src] * dinv[src]*dinv[dst]
__global__ void k_agg_edges(const long long* __restrict__ src, const long long* __restrict__ dst,
                            const float* __restrict__ t, const float* __restrict__ dinv,
                            float* __restrict__ out, int E) {
    int tid  = blockIdx.x * blockDim.x + threadIdx.x;
    int e    = tid >> 5;
    int lane = tid & 31;
    if (e >= E) return;
    int s = (int)src[e], d = (int)dst[e];
    float nrm = dinv[s] * dinv[d];
    const float4 v = ((const float4*)(t + (size_t)s * CH))[lane];
    float* od = out + (size_t)d * CH + lane * 4;
    unsafeAtomicAdd(od + 0, v.x * nrm);
    unsafeAtomicAdd(od + 1, v.y * nrm);
    unsafeAtomicAdd(od + 2, v.z * nrm);
    unsafeAtomicAdd(od + 3, v.w * nrm);
}

// ---------------- WMMA GEMM: C[M x 128] = A[M x 128] * B[128 x 128] ----------------
// One wave32 per 16-row block; computes all 8 column tiles so A is loaded once.
// bf16 inputs, fp32 accumulation via v_wmma_f32_16x16x32_bf16.
__global__ __launch_bounds__(32)
void k_gemm_bf16(const __bf16* __restrict__ A, const __bf16* __restrict__ B,
                 float* __restrict__ C, int M) {
    const int r0   = blockIdx.x * 16;
    const int lane = threadIdx.x & 31;
    const int m    = lane & 15;
    const int hi   = lane >> 4;            // lane group: 0 or 1

    int r = r0 + m;
    if (r >= M) r = M - 1;                 // clamp loads; stores guarded below
    const __bf16* arow = A + (size_t)r * CH;

    v8f acc[8] = {};

#pragma unroll
    for (int kc = 0; kc < 4; ++kc) {
        const int k0 = kc * 32;
        // A 16x32 bf16 per ISA layout: lane<16 -> K {0..7,16..23}, lane>=16 -> K {8..15,24..31}
        union { v16bf v; v8bf h[2]; } a;
        a.h[0] = *(const v8bf*)(arow + k0 + hi * 8);
        a.h[1] = *(const v8bf*)(arow + k0 + hi * 8 + 16);
#pragma unroll
        for (int n = 0; n < 8; ++n) {
            // B 32x16 bf16: lane L holds column N=L%16, K {0..7,16..23} (L<16) or {8..15,24..31} (L>=16)
            v16bf b;
#pragma unroll
            for (int s = 0; s < 16; ++s)
                b[s] = B[(size_t)(k0 + hi * 8 + (s & 7) + (s >> 3) * 16) * CH + n * 16 + m];
            acc[n] = __builtin_amdgcn_wmma_f32_16x16x32_bf16(
                false, a.v, false, b, (short)0, acc[n], false, false);
        }
    }

    // D layout: VGPR j -> M = j + 8*(lane>=16), N = lane%16
    const int n  = lane & 15;
    const int mb = hi * 8;
#pragma unroll
    for (int nt = 0; nt < 8; ++nt) {
#pragma unroll
        for (int j = 0; j < 8; ++j) {
            int rr = r0 + mb + j;
            if (rr < M) C[(size_t)rr * CH + nt * 16 + n] = acc[nt][j];
        }
    }
}

// ---------------- subgraph mean-pool + classifier ----------------

__global__ void k_pool(const float* __restrict__ h, const long long* __restrict__ sg,
                       float* __restrict__ emb, int S, int L) {
    int s = blockIdx.x;          // one block per subgraph
    int c = threadIdx.x;         // 128 channels
    float sum = 0.f; int cnt = 0;
    for (int j = 0; j < L; ++j) {
        long long idx = sg[(size_t)s * L + j];
        if (idx >= 0) { sum += h[(size_t)idx * CH + c]; cnt++; }
    }
    emb[(size_t)s * CH + c] = sum / (float)(cnt > 0 ? cnt : 1);
}

__global__ void k_cls(const float* __restrict__ emb, const float* __restrict__ Wc,
                      const float* __restrict__ bc, float* __restrict__ out,
                      int S, int NC) {
    int gid = blockIdx.x * blockDim.x + threadIdx.x;
    if (gid >= S * NC) return;
    int s = gid / NC, c = gid % NC;
    float acc = bc[c];
    const float* e = emb + (size_t)s * CH;
#pragma unroll 8
    for (int k = 0; k < CH; ++k) acc += e[k] * Wc[k * NC + c];
    out[gid] = acc;
}

// ---------------- launch ----------------

extern "C" void kernel_launch(void* const* d_in, const int* in_sizes, int n_in,
                              void* d_out, int out_size, void* d_ws, size_t ws_size,
                              hipStream_t stream) {
    const float*     x   = (const float*)d_in[0];
    const long long* ei  = (const long long*)d_in[1];
    const long long* sg  = (const long long*)d_in[2];
    const float*     W1  = (const float*)d_in[3];
    const float*     b1  = (const float*)d_in[4];
    const float*     W2  = (const float*)d_in[5];
    const float*     b2  = (const float*)d_in[6];
    const float*     Wc  = (const float*)d_in[7];
    const float*     bc  = (const float*)d_in[8];

    const int N  = in_sizes[0] / CH;     // 100000
    const int E  = in_sizes[1] / 2;      // 1.6M
    const int L  = 64;
    const int S  = in_sizes[2] / L;      // 4096
    const int NC = in_sizes[8];          // 10
    const int NF = N * CH;

    const long long* src = ei;
    const long long* dst = ei + E;

    // carve workspace (reused across layers)
    char* p = (char*)d_ws;
    auto carve = [&](size_t bytes) -> char* {
        char* r = p; p += (bytes + 255) & ~(size_t)255; return r;
    };
    float*  deg  = (float*)carve((size_t)N * 4);
    float*  dinv = (float*)carve((size_t)N * 4);
    __bf16* hb   = (__bf16*)carve((size_t)NF * 2);      // bf16 GEMM input (shared L1/L2)
    __bf16* w1b  = (__bf16*)carve((size_t)CH * CH * 2);
    __bf16* w2b  = (__bf16*)carve((size_t)CH * CH * 2);
    float*  t    = (float*)carve((size_t)NF * 4);       // GEMM output (reused)
    float*  out  = (float*)carve((size_t)NF * 4);       // aggregation output (reused)
    float*  emb  = (float*)carve((size_t)S * CH * 4);

    const int TH = 256;
    auto blks = [&](long long n) { return (int)((n + TH - 1) / TH); };

    // degrees (self-loop => start at 1), symmetric normalization
    k_fill<<<blks(N), TH, 0, stream>>>(deg, 1.0f, N);
    k_deg <<<blks(E), TH, 0, stream>>>(dst, deg, E);
    k_dinv<<<blks(N), TH, 0, stream>>>(deg, dinv, N);

    // bf16 copies of GEMM operands
    k_cvt<<<blks(NF),     TH, 0, stream>>>(x,  hb,  NF);
    k_cvt<<<blks(CH*CH),  TH, 0, stream>>>(W1, w1b, CH * CH);
    k_cvt<<<blks(CH*CH),  TH, 0, stream>>>(W2, w2b, CH * CH);

    // ---- layer 1: t = h @ W1 ; out = norm-aggregate(t) + b1 ; h = relu(out) ----
    k_gemm_bf16<<<(N + 15) / 16, 32, 0, stream>>>(hb, w1b, t, N);
    k_agg_init <<<blks(NF), TH, 0, stream>>>(out, t, dinv, b1, NF);
    k_agg_edges<<<blks((long long)E * 32), TH, 0, stream>>>(src, dst, t, dinv, out, E);
    k_relu_cvt <<<blks(NF), TH, 0, stream>>>(out, hb, NF);

    // ---- layer 2 (no relu) ----
    k_gemm_bf16<<<(N + 15) / 16, 32, 0, stream>>>(hb, w2b, t, N);
    k_agg_init <<<blks(NF), TH, 0, stream>>>(out, t, dinv, b2, NF);
    k_agg_edges<<<blks((long long)E * 32), TH, 0, stream>>>(src, dst, t, dinv, out, E);

    // ---- mean-pool over subgraphs, then classifier ----
    k_pool<<<S, CH, 0, stream>>>(out, sg, emb, S, L);
    k_cls <<<blks((long long)S * NC), TH, 0, stream>>>(emb, Wc, bc, (float*)d_out, S, NC);
}
